// BaseModel_87170656240449
// MI455X (gfx1250) — matrix operand order repaired and checked by reading, not compiled, not run on hardware
//
#include <hip/hip_runtime.h>
#include <stdint.h>

// CDNA5 wave32 WMMA types (probe-confirmed signatures)
typedef __attribute__((ext_vector_type(16))) __bf16 v16bf;
typedef __attribute__((ext_vector_type(8)))  float  v8f;

union FragAB { v16bf v; uint4 q[2]; };  // 32 bytes = 16 bf16 = 8 VGPRs

__device__ __forceinline__ unsigned short f32_to_bf16(float f) {
  union { float f; uint32_t u; } x; x.f = f;
  uint32_t u = x.u;
  if ((u & 0x7fffffffu) > 0x7f800000u)           // NaN: keep quiet
    return (unsigned short)((u >> 16) | 0x40u);
  return (unsigned short)((u + 0x7fffu + ((u >> 16) & 1u)) >> 16); // RNE
}

// C[M x 128] = A[M x K] @ W[K x 128], bf16 WMMA with f32 accumulate.
// Block = 256 threads = 8 waves; each wave owns one 16-col tile (8*16 = 128).
// Each block processes 4 row-tiles (64 rows), reusing the LDS-staged W panel.
template <int K>
__global__ __launch_bounds__(256) void gemm_bf16_wmma(const float* __restrict__ A,
                                                      const float* __restrict__ W,
                                                      float* __restrict__ C,
                                                      int M) {
  constexpr int PAD = 8;            // LDS bank-conflict padding (halves)
  constexpr int LDK = K + PAD;
  __shared__ unsigned short sW[128 * LDK];  // W transposed: [n][k] bf16
  __shared__ unsigned short sA[16 * LDK];   // A tile: [m][k] bf16

  const int tid  = threadIdx.x;
  const int lane = tid & 31;
  const int wave = tid >> 5;
  const int hi   = lane >> 4;       // half-wave selector (VGPR layout)
  const int lo   = lane & 15;

  // Stage full weight panel once: K x 128 f32 (row-major) -> sW[n][k] bf16.
  for (int idx = tid; idx < K * 128; idx += 256) {
    const int k = idx >> 7;         // row of W
    const int n = idx & 127;        // col of W
    sW[n * LDK + k] = f32_to_bf16(W[idx]);
  }

  for (int t = 0; t < 4; ++t) {
    const int r0 = (blockIdx.x * 4 + t) * 16;   // uniform across block
    if (r0 >= M) break;
    __syncthreads();                 // W visible (t==0) / prev tile done

    // Stage A rows [r0, r0+16) x K, f32 -> bf16.
    for (int idx = tid; idx < 16 * K; idx += 256) {
      const int m = idx / K;
      const int k = idx - m * K;
      const int row = r0 + m;
      const float v = (row < M) ? A[row * K + k] : 0.0f;
      sA[m * LDK + k] = f32_to_bf16(v);
    }
    __syncthreads();

    const int n0 = wave * 16;
    v8f c = {};
#pragma unroll
    for (int kc = 0; kc < K / 32; ++kc) {
      const int k0 = kc * 32;
      FragAB fa, fb;
      // A frag (16-bit A 16x32 layout): lane lo = M row;
      //   hi=0 -> K {0..7, 16..23}; hi=1 -> K {8..15, 24..31}
      fa.q[0] = *reinterpret_cast<const uint4*>(&sA[lo * LDK + k0 + 8 * hi]);
      fa.q[1] = *reinterpret_cast<const uint4*>(&sA[lo * LDK + k0 + 16 + 8 * hi]);
      // B frag: lane lo = N col; hi=0 -> K {0..7, 16..23}, hi=1 -> K {8..15, 24..31} (sW is [n][k])
      const unsigned short* pb = &sW[(n0 + lo) * LDK + k0 + 8 * hi];
      fb.q[0] = *reinterpret_cast<const uint4*>(pb);
      fb.q[1] = *reinterpret_cast<const uint4*>(pb + 16);
      c = __builtin_amdgcn_wmma_f32_16x16x32_bf16(false, fa.v, false, fb.v,
                                                  (short)0, c, false, false);
    }

    // C/D layout: VGPR r, lanes 0-15 -> M=r, lanes 16-31 -> M=r+8; N = lo.
#pragma unroll
    for (int r = 0; r < 8; ++r) {
      const int row = r0 + r + 8 * hi;
      if (row < M) C[row * 128 + n0 + lo] = c[r];
    }
  }
}

// COO scatter: out[rows[e], :] += vals[e] * S[cols[e], :]
// 128 consecutive threads share one edge -> coalesced row reads + atomics.
__global__ __launch_bounds__(256) void spmm_scatter(const float* __restrict__ vals,
                                                    const int* __restrict__ rows,
                                                    const int* __restrict__ cols,
                                                    const float* __restrict__ S,
                                                    float* __restrict__ out,
                                                    int nE) {
  const long long t = (long long)blockIdx.x * 256 + threadIdx.x;
  const int e = (int)(t >> 7);
  const int f = (int)(t & 127);
  if (e >= nE) return;
  const int r = rows[e];
  const int c = cols[e];
  const float v = vals[e];
  atomicAdd(&out[(long long)r * 128 + f], v * S[(long long)c * 128 + f]);
}

__global__ __launch_bounds__(256) void bias_relu_inplace(float* __restrict__ x,
                                                         const float* __restrict__ b,
                                                         int n) {
  const int t = blockIdx.x * 256 + threadIdx.x;
  if (t < n) {
    const float y = x[t] + b[t & 127];
    x[t] = y > 0.0f ? y : 0.0f;
  }
}

__global__ __launch_bounds__(256) void zero_f4(float4* __restrict__ p, int n4) {
  const int t = blockIdx.x * 256 + threadIdx.x;
  if (t < n4) p[t] = make_float4(0.f, 0.f, 0.f, 0.f);
}

extern "C" void kernel_launch(void* const* d_in, const int* in_sizes, int n_in,
                              void* d_out, int out_size, void* d_ws, size_t ws_size,
                              hipStream_t stream) {
  const float* feat = (const float*)d_in[0];   // (M, 256) f32
  const int*   rows = (const int*)  d_in[1];   // (E,) i32
  const int*   cols = (const int*)  d_in[2];   // (E,) i32
  const float* vals = (const float*)d_in[3];   // (E,) f32
  const float* W1   = (const float*)d_in[4];   // (256, 128) f32
  const float* b1   = (const float*)d_in[5];   // (128,) f32
  const float* W2   = (const float*)d_in[6];   // (128, 128) f32
  const float* b2   = (const float*)d_in[7];   // (128,) f32
  float* out = (float*)d_out;                  // (M, 128) f32

  const int M = in_sizes[0] / 256;             // 50000
  const int E = in_sizes[1];                   // 800000
  const long long HN = (long long)M * 128;

  float* S = (float*)d_ws;                     // support buffer (M x 128)
  float* H = S + HN;                           // aggregate / hidden (M x 128)

  const dim3 blk(256);
  const int gGemm = (M + 63) / 64;
  const int gZero = (int)((HN / 4 + 255) / 256);
  const int gBias = (int)((HN + 255) / 256);
  const int gSpmm = (int)(((long long)E * 128 + 255) / 256);

  // Layer 1: S = feat @ W1 ; H = relu(A*S + b1)
  gemm_bf16_wmma<256><<<gGemm, blk, 0, stream>>>(feat, W1, S, M);
  zero_f4<<<gZero, blk, 0, stream>>>((float4*)H, (int)(HN / 4));
  spmm_scatter<<<gSpmm, blk, 0, stream>>>(vals, rows, cols, S, H, E);
  bias_relu_inplace<<<gBias, blk, 0, stream>>>(H, b1, (int)HN);

  // Layer 2: S = H @ W2 ; out = relu(A*S + b2)
  gemm_bf16_wmma<128><<<gGemm, blk, 0, stream>>>(H, W2, S, M);
  zero_f4<<<gZero, blk, 0, stream>>>((float4*)out, (int)(HN / 4));
  spmm_scatter<<<gSpmm, blk, 0, stream>>>(vals, rows, cols, S, out, E);
  bias_relu_inplace<<<gBias, blk, 0, stream>>>(out, b2, (int)HN);
}
